// ImplicitAttetionLayer_52106543235234
// MI455X (gfx1250) — matrix operand order repaired and checked by reading, not compiled, not run on hardware
//
#include <hip/hip_runtime.h>
#include <hip/hip_bf16.h>

// ---------------------------------------------------------------------------
// ImplicitAttentionLayer for MI455X (gfx1250, wave32, WMMA bf16 16x16x32)
//
// out[m,o] = sum_{h,k} (attn[m,h]*x[m,k]) * Wv[h,o,k] + sum_h attn[m,h]*bv[h,o]
//   -> ONE dense GEMM with K_eff = H*IN = 8192, A_eff formed on the fly.
// attn computed first by a small WMMA kernel into d_ws.
//
// Round-3 change: f32->bf16 packing via inline-asm v_cvt_pk_bf16_f32 so SLP
// cannot re-pair the conversions and insert v_perm_b32 interleaves.
// ---------------------------------------------------------------------------

#define N_HEADS 8
#define IN_DIM 1024
#define ATTN_DIM 128
#define OUT_DIM 1024
#define BATCH 32768

typedef __attribute__((ext_vector_type(8)))  float  v8f;
typedef __attribute__((ext_vector_type(16))) __bf16 v16bf;

union Frag {
    v16bf v;
    uint4 q[2];
    unsigned short s[16];
};

// Two f32 -> one packed-bf16 dword, exactly one VALU op, pairing fixed.
static __device__ __forceinline__ unsigned int pack2(float a, float b) {
    unsigned int u;
    asm("v_cvt_pk_bf16_f32 %0, %1, %2" : "=v"(u) : "v"(a), "v"(b));
    return u;
}

// B fragment (32x16 bf16, K x N): lane%16 = column n, lane/16 selects K 0..15
// or K 16..31; each lane reads 16 CONSECUTIVE K values of one weight row.
static __device__ __forceinline__ void load_bfrag(const float* __restrict__ p, Frag& f) {
    const float4* p4 = (const float4*)p;
    float4 a = p4[0], b = p4[1], c = p4[2], d = p4[3];
    f.q[0] = make_uint4(pack2(a.x, a.y), pack2(a.z, a.w), pack2(b.x, b.y), pack2(b.z, b.w));
    f.q[1] = make_uint4(pack2(c.x, c.y), pack2(c.z, c.w), pack2(d.x, d.y), pack2(d.z, d.w));
}

static __device__ __forceinline__ v8f wmma_bf16(const Frag& a, const Frag& b, v8f c) {
    return __builtin_amdgcn_wmma_f32_16x16x32_bf16(false, a.v, false, b.v,
                                                   (short)0, c, false, false);
}

// ---------------------------------------------------------------------------
// Kernel 1: attention gates. One block = 16 batch rows, 4 waves.
// Wave w handles a-tiles 2w, 2w+1 (8 tiles of 16 cover ATTN_DIM=128).
// ---------------------------------------------------------------------------
#define K1_STRIDE_U 516   // (1024/2 + 4) dwords: 516 % 64 = 4 -> bank-conflict free

__global__ __launch_bounds__(128) void attn_gate_kernel(
    const float* __restrict__ x,  const float* __restrict__ Wq,
    const float* __restrict__ bq, const float* __restrict__ Wk,
    const float* __restrict__ bk, float* __restrict__ attn)
{
    __shared__ unsigned int ldsx[16 * K1_STRIDE_U];
    __shared__ float ldsdot[16 * 8];

    const int tid = threadIdx.x;
    const int b0  = blockIdx.x * 16;

    // Stage x[16 x 1024] fp32 -> bf16 LDS (read x from HBM exactly once here)
    for (int i = tid; i < 16 * 256; i += 128) {
        const int row = i >> 8, c4 = i & 255;
        const float4 f = *(const float4*)(x + (size_t)(b0 + row) * IN_DIM + c4 * 4);
        ldsx[row * K1_STRIDE_U + c4 * 2]     = pack2(f.x, f.y);
        ldsx[row * K1_STRIDE_U + c4 * 2 + 1] = pack2(f.z, f.w);
    }
    if (tid < 128) ldsdot[tid] = 0.0f;   // 16 rows x 8 heads
    __syncthreads();

    const int lane = tid & 31, wave = tid >> 5;
    const int ln = lane & 15, hi = lane >> 4;

    v8f dp[N_HEADS] = {};   // per-lane partial q.k for the 8 rows this lane holds

    #pragma unroll
    for (int t = 0; t < 2; ++t) {
        const int a0 = (wave * 2 + t) * 16;
        v8f qacc = {};
        v8f kacc[N_HEADS] = {};
        for (int k0 = 0; k0 < IN_DIM; k0 += 32) {
            // A fragment (16x32 bf16): lane half picks K 0..15 or K 16..31 (contiguous)
            Frag A;
            const uint4* pa = (const uint4*)&ldsx[ln * K1_STRIDE_U + (k0 >> 1) + hi * 8];
            A.q[0] = pa[0];
            A.q[1] = pa[1];  // +4 dwords = +8 bf16

            Frag Bq;
            load_bfrag(Wq + (size_t)(a0 + ln) * IN_DIM + k0 + hi * 16, Bq);
            qacc = wmma_bf16(A, Bq, qacc);

            #pragma unroll
            for (int h = 0; h < N_HEADS; ++h) {
                Frag Bk;
                load_bfrag(Wk + ((size_t)h * ATTN_DIM + a0 + ln) * IN_DIM + k0 + hi * 16, Bk);
                kacc[h] = wmma_bf16(A, Bk, kacc[h]);
            }
        }
        // C/D layout: lane holds column a0+ln, rows r (+8 for upper half)
        const float qb = bq[a0 + ln];
        #pragma unroll
        for (int h = 0; h < N_HEADS; ++h) {
            const float kb = bk[h * ATTN_DIM + a0 + ln];
            #pragma unroll
            for (int r = 0; r < 8; ++r)
                dp[h][r] += (qacc[r] + qb) * (kacc[h][r] + kb);
        }
    }

    // Butterfly-reduce over the 16 columns held by each half-wave (wave32)
    #pragma unroll
    for (int m = 1; m < 16; m <<= 1)
        #pragma unroll
        for (int h = 0; h < N_HEADS; ++h)
            #pragma unroll
            for (int r = 0; r < 8; ++r)
                dp[h][r] += __shfl_xor(dp[h][r], m, 32);

    // Accumulate across the 4 waves via LDS float atomics (ds_add_f32)
    if (ln == 0) {
        #pragma unroll
        for (int h = 0; h < N_HEADS; ++h)
            #pragma unroll
            for (int r = 0; r < 8; ++r)
                atomicAdd(&ldsdot[(r + hi * 8) * 8 + h], dp[h][r]);
    }
    __syncthreads();

    // Softmax over heads, one thread per batch row
    if (tid < 16) {
        const float scale = 0.088388347648318447f;  // 1/sqrt(128)
        float d[N_HEADS], mx = -3.0e38f;
        #pragma unroll
        for (int h = 0; h < N_HEADS; ++h) {
            d[h] = ldsdot[tid * 8 + h] * scale;
            mx = fmaxf(mx, d[h]);
        }
        float s = 0.0f;
        #pragma unroll
        for (int h = 0; h < N_HEADS; ++h) { d[h] = __expf(d[h] - mx); s += d[h]; }
        const float inv = 1.0f / s;
        #pragma unroll
        for (int h = 0; h < N_HEADS; ++h)
            attn[(size_t)(b0 + tid) * N_HEADS + h] = d[h] * inv;
    }
}

// ---------------------------------------------------------------------------
// Kernel 2: fused output GEMM, K_eff = 8192 (heads folded into K).
// Block = 256 threads (8 waves), out tile 128x128; wave = 2x4 WMMA tiles.
// LDS stages A_eff = attn*x and B = Wv, 128x64 bf16 each, stride 36 dwords
// (36*r mod 64 distinct for r<16 -> conflict-free fragment reads).
// ---------------------------------------------------------------------------
#define K2_BM 128
#define K2_BN 128
#define K2_BK 64
#define K2_STRIDE_U 36

__global__ __launch_bounds__(256) void attn_out_kernel(
    const float* __restrict__ x,  const float* __restrict__ Wv,
    const float* __restrict__ bv, const float* __restrict__ attn,
    float* __restrict__ out)
{
    __shared__ unsigned int ldsA[K2_BM * K2_STRIDE_U];
    __shared__ unsigned int ldsB[K2_BN * K2_STRIDE_U];
    __shared__ float ldsattn[K2_BM * N_HEADS];

    const int tid = threadIdx.x;
    const int b0  = blockIdx.x * K2_BM;
    const int o0  = blockIdx.y * K2_BN;

    // Stage attn[128 x 8] once
    for (int i = tid; i < K2_BM * N_HEADS; i += 256)
        ldsattn[i] = attn[(size_t)b0 * N_HEADS + i];
    __syncthreads();

    const int lane = tid & 31, wave = tid >> 5;
    const int wm = wave >> 1, wn = wave & 1;      // 4x2 wave grid
    const int ln = lane & 15, hi = lane >> 4;

    // Staging coords: 2 threads per row, each moves 32 consecutive floats
    const int srow = tid >> 1;
    const int scol = (tid & 1) * 32;

    v8f acc[2][4] = {};

    for (int h = 0; h < N_HEADS; ++h) {
        const float* wvh = Wv + (size_t)h * OUT_DIM * IN_DIM;
        for (int k0 = 0; k0 < IN_DIM; k0 += K2_BK) {
            __syncthreads();
            // Stage A_eff = attn[row,h] * x[row, k0+..] as bf16
            {
                const float sc = ldsattn[srow * N_HEADS + h];
                const float4* xs =
                    (const float4*)(x + (size_t)(b0 + srow) * IN_DIM + k0 + scol);
                #pragma unroll
                for (int j = 0; j < 4; ++j) {
                    float4 fa = xs[2 * j], fb = xs[2 * j + 1];
                    *(uint4*)&ldsA[srow * K2_STRIDE_U + (tid & 1) * 16 + j * 4] =
                        make_uint4(pack2(fa.x * sc, fa.y * sc), pack2(fa.z * sc, fa.w * sc),
                                   pack2(fb.x * sc, fb.y * sc), pack2(fb.z * sc, fb.w * sc));
                }
            }
            // Stage B = Wv[h][o0+row][k0+..] as bf16
            {
                const float4* ws =
                    (const float4*)(wvh + (size_t)(o0 + srow) * IN_DIM + k0 + scol);
                #pragma unroll
                for (int j = 0; j < 4; ++j) {
                    float4 fa = ws[2 * j], fb = ws[2 * j + 1];
                    *(uint4*)&ldsB[srow * K2_STRIDE_U + (tid & 1) * 16 + j * 4] =
                        make_uint4(pack2(fa.x, fa.y), pack2(fa.z, fa.w),
                                   pack2(fb.x, fb.y), pack2(fb.z, fb.w));
                }
            }
            __syncthreads();

            #pragma unroll
            for (int ks = 0; ks < K2_BK / 32; ++ks) {
                Frag Afr[2];
                #pragma unroll
                for (int mt = 0; mt < 2; ++mt) {
                    const int row = wm * 32 + mt * 16 + ln;
                    const uint4* pa =
                        (const uint4*)&ldsA[row * K2_STRIDE_U + ks * 16 + hi * 8];
                    Afr[mt].q[0] = pa[0];
                    Afr[mt].q[1] = pa[1];
                }
                #pragma unroll
                for (int nt = 0; nt < 4; ++nt) {
                    const int n = wn * 64 + nt * 16 + ln;
                    Frag Bfr;
                    const uint4* pb =
                        (const uint4*)&ldsB[n * K2_STRIDE_U + ks * 16 + hi * 8];
                    Bfr.q[0] = pb[0];
                    Bfr.q[1] = pb[1];
                    #pragma unroll
                    for (int mt = 0; mt < 2; ++mt)
                        acc[mt][nt] = wmma_bf16(Afr[mt], Bfr, acc[mt][nt]);
                }
            }
        }
    }

    // Epilogue: add attn-weighted bias, scatter f32 rows
    #pragma unroll
    for (int nt = 0; nt < 4; ++nt) {
        const int o = o0 + wn * 64 + nt * 16 + ln;
        float bvv[N_HEADS];
        #pragma unroll
        for (int h = 0; h < N_HEADS; ++h) bvv[h] = bv[h * OUT_DIM + o];
        #pragma unroll
        for (int mt = 0; mt < 2; ++mt) {
            #pragma unroll
            for (int r = 0; r < 8; ++r) {
                const int ml = wm * 32 + mt * 16 + r + hi * 8;
                float s = acc[mt][nt][r];
                #pragma unroll
                for (int h = 0; h < N_HEADS; ++h)
                    s += ldsattn[ml * N_HEADS + h] * bvv[h];
                out[(size_t)(b0 + ml) * OUT_DIM + o] = s;
            }
        }
    }
}

// ---------------------------------------------------------------------------
extern "C" void kernel_launch(void* const* d_in, const int* in_sizes, int n_in,
                              void* d_out, int out_size, void* d_ws, size_t ws_size,
                              hipStream_t stream) {
    const float* x  = (const float*)d_in[0];
    const float* Wq = (const float*)d_in[1];
    const float* bq = (const float*)d_in[2];
    const float* Wk = (const float*)d_in[3];
    const float* bk = (const float*)d_in[4];
    const float* Wv = (const float*)d_in[5];
    const float* bv = (const float*)d_in[6];
    float* out  = (float*)d_out;
    float* attn = (float*)d_ws;   // BATCH * N_HEADS floats = 1 MB

    attn_gate_kernel<<<dim3(BATCH / 16), 128, 0, stream>>>(x, Wq, bq, Wk, bk, attn);
    attn_out_kernel<<<dim3(BATCH / K2_BM, OUT_DIM / K2_BN), 256, 0, stream>>>(
        x, Wv, bv, attn, out);
}
